// MultiHeadAttention_77266461655215
// MI455X (gfx1250) — matrix operand order repaired and checked by reading, not compiled, not run on hardware
//
#include <hip/hip_runtime.h>
#include <hip/hip_bf16.h>
#include <stddef.h>

// ---------------------------------------------------------------------------
// MHA forward for MI455X (gfx1250, wave32, WMMA).
//   B=4, L=2048, D=1024, H=16, dk=64
// Bandwidth analysis: mandatory traffic is dominated by the fp32 attn_weights
// tensor (1.07 GB write + lower-triangle re-read ~0.54 GB). All GEMMs run on
// v_wmma_f32_16x16x32_bf16 with fp32 accumulate; softmax and P stay fp32.
// Causal structure is exploited: upper-triangle score tiles are never
// computed (written as exact 0), and the PV GEMM only reads k <= row.
// 32x32 register blocking (4 WMMAs / 8 b128 loads) in the big GEMMs.
// Q strips are staged to LDS via global_load_async_to_lds_b128 (ASYNCcnt).
// ---------------------------------------------------------------------------

typedef __attribute__((ext_vector_type(16))) __bf16 v16bf;
typedef __attribute__((ext_vector_type(8)))  __bf16 v8bf;
typedef __attribute__((ext_vector_type(8)))  float  v8f;
typedef __attribute__((ext_vector_type(4)))  float  v4f;

#define BATCH 4
#define SEQ   2048
#define DMODEL 1024
#define NHEAD 16
#define DK    64
#define BL    (BATCH*SEQ)        // 8192 rows
#define BH    (BATCH*NHEAD)      // 64 (b,h) pairs

// ---- WMMA helpers ---------------------------------------------------------

__device__ __forceinline__ v8f wmma_bf16(v16bf a, v16bf b, v8f c) {
  return __builtin_amdgcn_wmma_f32_16x16x32_bf16(false, a, false, b,
                                                 (short)0, c, false, false);
}

// A-matrix 16x32 bf16 fragment (lanes 0-15: M=0..15 K={0..7,16..23};
// lanes 16-31: M=0..15 K={8..15,24..31}).
__device__ __forceinline__ v16bf load_a_bf16(const __bf16* base,
                                             int ldk, int lane, int kbase) {
  const int m  = lane & 15;
  const int hi = lane >> 4;
  const __bf16* p0 = base + (size_t)m * ldk + kbase + hi * 8;
  const __bf16* p1 = base + (size_t)m * ldk + kbase + 16 + hi * 8;
  v8bf lo = *(const v8bf*)p0;
  v8bf hh = *(const v8bf*)p1;
  v16bf a;
#pragma unroll
  for (int i = 0; i < 8; ++i) { a[i] = lo[i]; a[8 + i] = hh[i]; }
  return a;
}

// Same fragment but fp32 source (convert on load) — used for P (attn probs).
__device__ __forceinline__ v16bf load_a_f32(const float* __restrict__ base,
                                            int ldk, int lane, int kbase) {
  const int m  = lane & 15;
  const int hi = lane >> 4;
  const float* p0 = base + (size_t)m * ldk + kbase + hi * 8;
  const float* p1 = base + (size_t)m * ldk + kbase + 16 + hi * 8;
  v16bf a;
#pragma unroll
  for (int i = 0; i < 8; ++i) { a[i] = (__bf16)p0[i]; a[8 + i] = (__bf16)p1[i]; }
  return a;
}

// B-matrix 32x16 bf16 fragment: lane's column n = lane&15, K values
// kbase + (lane>>4)*8 + {0..7, 16..23}; "base" rows are B columns (rows of W/K/V^T).
__device__ __forceinline__ v16bf load_b_bf16(const __bf16* __restrict__ base,
                                             int ldk, int lane, int kbase) {
  const int n  = lane & 15;
  const int hi = lane >> 4;
  const __bf16* p = base + (size_t)n * ldk + kbase + hi * 8;
  v8bf lo = *(const v8bf*)p;
  v8bf hh = *(const v8bf*)(p + 16);
  v16bf b;
#pragma unroll
  for (int i = 0; i < 8; ++i) { b[i] = lo[i]; b[8 + i] = hh[i]; }
  return b;
}

// ---- Stage 0: fp32 -> bf16 conversion -------------------------------------

__global__ __launch_bounds__(256) void f32_to_bf16_kernel(
    const float* __restrict__ src, __bf16* __restrict__ dst, int n4) {
  int i = blockIdx.x * blockDim.x + threadIdx.x;
  if (i < n4) {
    v4f v = *(const v4f*)(src + (size_t)i * 4);
    __bf16* d = dst + (size_t)i * 4;
    d[0] = (__bf16)v.x; d[1] = (__bf16)v.y; d[2] = (__bf16)v.z; d[3] = (__bf16)v.w;
  }
}

// ---- Stage 1: fused QKV projection (32x32 tile per wave) ------------------
// Y = x @ W^T + b.  Q,K stored [BH, L, dk] bf16; V stored transposed
// [BH, dk, L] bf16 so PV B-fragments are contiguous along K.

__device__ __forceinline__ void store_qkv_subtile(
    v8f c, int mat, int m0, int n0, int lane, const float* __restrict__ bias,
    __bf16* __restrict__ qout, __bf16* __restrict__ kout,
    __bf16* __restrict__ vtout) {
  const int n  = n0 + (lane & 15);
  const int hi = lane >> 4;
  const float bb = bias[n];
  const int h = n >> 6, d = n & 63;
#pragma unroll
  for (int i = 0; i < 8; ++i) {
    const int m    = m0 + i + 8 * hi;
    const int bidx = m >> 11;
    const int l    = m & (SEQ - 1);
    const int bh   = bidx * NHEAD + h;
    const __bf16 val = (__bf16)(c[i] + bb);
    if (mat == 2)      vtout[((size_t)bh * DK + d) * SEQ + l] = val;
    else if (mat == 0) qout[((size_t)bh * SEQ + l) * DK + d] = val;
    else               kout[((size_t)bh * SEQ + l) * DK + d] = val;
  }
}

__global__ __launch_bounds__(256) void qkv_gemm_kernel(
    const __bf16* __restrict__ xb,
    const __bf16* __restrict__ wq, const __bf16* __restrict__ wk,
    const __bf16* __restrict__ wv,
    const float* __restrict__ bq, const float* __restrict__ bk,
    const float* __restrict__ bv,
    __bf16* __restrict__ qout, __bf16* __restrict__ kout,
    __bf16* __restrict__ vtout) {
  const int lane = threadIdx.x & 31;
  const int gw   = blockIdx.x * 8 + (threadIdx.x >> 5);
  const int mat  = gw >> 13;                 // 8192 32x32 tiles per matrix
  const int rem  = gw & 8191;
  const int m0   = (rem >> 5) * 32;          // 256 M-tiles
  const int n0   = (rem & 31) * 32;          // 32  N-tiles
  const __bf16* W    = (mat == 0) ? wq : (mat == 1) ? wk : wv;
  const float*  bias = (mat == 0) ? bq : (mat == 1) ? bk : bv;

  const __bf16* a0b = xb + (size_t)m0 * DMODEL;
  const __bf16* a1b = a0b + (size_t)16 * DMODEL;
  const __bf16* b0b = W  + (size_t)n0 * DMODEL;
  const __bf16* b1b = b0b + (size_t)16 * DMODEL;
  v8f c00 = {}, c01 = {}, c10 = {}, c11 = {};
  for (int k = 0; k < DMODEL; k += 32) {
    v16bf a0 = load_a_bf16(a0b, DMODEL, lane, k);
    v16bf a1 = load_a_bf16(a1b, DMODEL, lane, k);
    v16bf b0 = load_b_bf16(b0b, DMODEL, lane, k);
    v16bf b1 = load_b_bf16(b1b, DMODEL, lane, k);
    c00 = wmma_bf16(a0, b0, c00);
    c01 = wmma_bf16(a0, b1, c01);
    c10 = wmma_bf16(a1, b0, c10);
    c11 = wmma_bf16(a1, b1, c11);
  }
  store_qkv_subtile(c00, mat, m0,      n0,      lane, bias, qout, kout, vtout);
  store_qkv_subtile(c01, mat, m0,      n0 + 16, lane, bias, qout, kout, vtout);
  store_qkv_subtile(c10, mat, m0 + 16, n0,      lane, bias, qout, kout, vtout);
  store_qkv_subtile(c11, mat, m0 + 16, n0 + 16, lane, bias, qout, kout, vtout);
}

// ---- Stage 2: scores + causal mask + softmax ------------------------------
// One block per (bh, 16-row q-strip).  Q strip (16x64 bf16 = 2KB) is staged
// into LDS with global_load_async_to_lds_b128; only the lower-triangle column
// tiles (ct <= qt) are computed.  16x2048 fp32 scores live in 128KB dynamic
// LDS; columns beyond the diagonal tile are written as exact 0.

__global__ __launch_bounds__(256) void scores_softmax_kernel(
    const __bf16* __restrict__ qb, const __bf16* __restrict__ kb,
    float* __restrict__ pout) {
  extern __shared__ float smem[];                 // [16][2048] scores
  __shared__ __bf16 qtile[16 * DK];               // staged Q strip (2KB)
  __shared__ float red[16][17];
  __shared__ float rowstat[16];
  const int tid  = threadIdx.x;
  const int lane = tid & 31;
  const int w    = tid >> 5;
  const int bh   = blockIdx.x >> 7;               // 128 strips per (b,h)
  const int qt   = blockIdx.x & 127;
  const int m0   = qt * 16;
  const int colmax = m0 + 16;                     // first all-zero column

  const __bf16* qbase = qb + ((size_t)bh * SEQ + m0) * DK;
  const __bf16* kbhb  = kb + (size_t)bh * SEQ * DK;

  // Async-copy the 2KB Q strip into LDS: 128 lanes x 16B (ASYNCcnt-tracked).
  if (tid < 128) {
    unsigned lds_dst = (unsigned)(unsigned long long)(&qtile[0]) + tid * 16;
    const void* gsrc = (const char*)qbase + tid * 16;
    asm volatile("global_load_async_to_lds_b128 %0, %1, off"
                 :: "v"(lds_dst), "v"(gsrc) : "memory");
  }
  asm volatile("s_wait_asynccnt 0x0" ::: "memory");
  __syncthreads();

  // Hoisted A-fragments from LDS (shared by all column tiles of this wave).
  const v16bf aQ0 = load_a_bf16(qtile, DK, lane, 0);
  const v16bf aQ1 = load_a_bf16(qtile, DK, lane, 32);

  for (int ct = w; ct <= qt; ct += 8) {           // lower triangle only
    const __bf16* bbase = kbhb + (size_t)ct * 16 * DK;
    v8f c = {};
    c = wmma_bf16(aQ0, load_b_bf16(bbase, DK, lane, 0),  c);
    c = wmma_bf16(aQ1, load_b_bf16(bbase, DK, lane, 32), c);
    const int hi  = lane >> 4;
    const int col = ct * 16 + (lane & 15);
#pragma unroll
    for (int i = 0; i < 8; ++i) {
      const int r    = i + 8 * hi;
      const int grow = m0 + r;
      float v = c[i] * 0.125f;                    // 1/sqrt(64)
      if (col > grow) v = -__builtin_inff();      // causal (tril) mask
      smem[r * SEQ + col] = v;
    }
  }
  __syncthreads();

  // fp32 softmax over [0, colmax): 16 threads per row, stride-16 columns.
  const int r  = tid >> 4;
  const int cl = tid & 15;
  float mx = -__builtin_inff();
  for (int col = cl; col < colmax; col += 16) mx = fmaxf(mx, smem[r * SEQ + col]);
  red[r][cl] = mx;
  __syncthreads();
  if (cl == 0) {
    float m2 = red[r][0];
#pragma unroll
    for (int j = 1; j < 16; ++j) m2 = fmaxf(m2, red[r][j]);
    rowstat[r] = m2;
  }
  __syncthreads();
  const float rmax = rowstat[r];
  float sum = 0.f;
  for (int col = cl; col < colmax; col += 16) {
    float e = __expf(smem[r * SEQ + col] - rmax);
    smem[r * SEQ + col] = e;
    sum += e;
  }
  __syncthreads();
  red[r][cl] = sum;
  __syncthreads();
  if (cl == 0) {
    float s2 = 0.f;
#pragma unroll
    for (int j = 0; j < 16; ++j) s2 += red[r][j];
    rowstat[r] = 1.0f / s2;
  }
  __syncthreads();
  const float inv = rowstat[r];
  float* prow = pout + ((size_t)bh * SEQ + m0 + r) * SEQ;
  for (int col = cl; col < SEQ; col += 16)
    prow[col] = (col < colmax) ? smem[r * SEQ + col] * inv : 0.0f;
}

// ---- Stage 3: PV GEMM (32x32 tile per wave, lower-triangle K range) -------

__global__ __launch_bounds__(256) void av_gemm_kernel(
    const float* __restrict__ p, const __bf16* __restrict__ vt,
    __bf16* __restrict__ ao) {
  const int lane = threadIdx.x & 31;
  const int gw   = blockIdx.x * 8 + (threadIdx.x >> 5);
  const int bh   = gw >> 7;                  // 128 32x32 tiles per (b,h)
  const int rem  = gw & 127;
  const int m0   = (rem >> 1) * 32;          // 64 M-tiles
  const int n0   = (rem & 1) * 32;           // 2  N-tiles
  const int kmax = m0 + 32;                  // P[m,k]==0 for k>m
  const float*  a0b = p  + ((size_t)bh * SEQ + m0) * SEQ;
  const float*  a1b = a0b + (size_t)16 * SEQ;
  const __bf16* b0b = vt + ((size_t)bh * DK + n0) * SEQ;
  const __bf16* b1b = b0b + (size_t)16 * SEQ;
  v8f c00 = {}, c01 = {}, c10 = {}, c11 = {};
  for (int k = 0; k < kmax; k += 32) {
    v16bf a0 = load_a_f32(a0b, SEQ, lane, k);
    v16bf a1 = load_a_f32(a1b, SEQ, lane, k);
    v16bf b0 = load_b_bf16(b0b, SEQ, lane, k);
    v16bf b1 = load_b_bf16(b1b, SEQ, lane, k);
    c00 = wmma_bf16(a0, b0, c00);
    c01 = wmma_bf16(a0, b1, c01);
    c10 = wmma_bf16(a1, b0, c10);
    c11 = wmma_bf16(a1, b1, c11);
  }
  const int b_ = bh >> 4, h = bh & 15;
  const int hi = lane >> 4;
#pragma unroll
  for (int s = 0; s < 4; ++s) {
    const v8f c  = (s == 0) ? c00 : (s == 1) ? c01 : (s == 2) ? c10 : c11;
    const int mm = m0 + ((s >> 1) ? 16 : 0);
    const int n  = n0 + ((s & 1) ? 16 : 0) + (lane & 15);
#pragma unroll
    for (int i = 0; i < 8; ++i) {
      const int l = mm + i + 8 * hi;
      ao[((size_t)b_ * SEQ + l) * DMODEL + h * DK + n] = (__bf16)c[i];
    }
  }
}

// ---- Stage 4: output projection (32x32 tile per wave, fp32 out) -----------

__global__ __launch_bounds__(256) void oproj_gemm_kernel(
    const __bf16* __restrict__ ao, const __bf16* __restrict__ wo,
    const float* __restrict__ bo, float* __restrict__ out) {
  const int lane = threadIdx.x & 31;
  const int gw   = blockIdx.x * 8 + (threadIdx.x >> 5);
  const int m0   = (gw >> 5) * 32;           // 256 M-tiles
  const int n0   = (gw & 31) * 32;           // 32  N-tiles
  const __bf16* a0b = ao + (size_t)m0 * DMODEL;
  const __bf16* a1b = a0b + (size_t)16 * DMODEL;
  const __bf16* b0b = wo + (size_t)n0 * DMODEL;
  const __bf16* b1b = b0b + (size_t)16 * DMODEL;
  v8f c00 = {}, c01 = {}, c10 = {}, c11 = {};
  for (int k = 0; k < DMODEL; k += 32) {
    v16bf a0 = load_a_bf16(a0b, DMODEL, lane, k);
    v16bf a1 = load_a_bf16(a1b, DMODEL, lane, k);
    v16bf b0 = load_b_bf16(b0b, DMODEL, lane, k);
    v16bf b1 = load_b_bf16(b1b, DMODEL, lane, k);
    c00 = wmma_bf16(a0, b0, c00);
    c01 = wmma_bf16(a0, b1, c01);
    c10 = wmma_bf16(a1, b0, c10);
    c11 = wmma_bf16(a1, b1, c11);
  }
  const int hi = lane >> 4;
#pragma unroll
  for (int s = 0; s < 4; ++s) {
    const v8f c  = (s == 0) ? c00 : (s == 1) ? c01 : (s == 2) ? c10 : c11;
    const int mm = m0 + ((s >> 1) ? 16 : 0);
    const int n  = n0 + ((s & 1) ? 16 : 0) + (lane & 15);
    const float bias = bo[n];
#pragma unroll
    for (int i = 0; i < 8; ++i) {
      const int m = mm + i + 8 * hi;
      out[(size_t)m * DMODEL + n] = c[i] + bias;
    }
  }
}

// ---- Host side ------------------------------------------------------------

extern "C" void kernel_launch(void* const* d_in, const int* in_sizes, int n_in,
                              void* d_out, int out_size, void* d_ws, size_t ws_size,
                              hipStream_t stream) {
  const float* x  = (const float*)d_in[0];
  // d_in[1] = mask (tril) — causal structure applied analytically in-kernel.
  const float* Wq = (const float*)d_in[2];
  const float* bq = (const float*)d_in[3];
  const float* Wk = (const float*)d_in[4];
  const float* bk = (const float*)d_in[5];
  const float* Wv = (const float*)d_in[6];
  const float* bv = (const float*)d_in[7];
  const float* Wo = (const float*)d_in[8];
  const float* bo = (const float*)d_in[9];

  float* out   = (float*)d_out;                       // [B,L,D]
  float* pattn = out + (size_t)BL * DMODEL;           // [B,H,L,L] fp32

  char* ws = (char*)d_ws;
  const size_t XB  = (size_t)BL * DMODEL * 2;         // 16 MiB
  const size_t WB  = (size_t)DMODEL * DMODEL * 2;     //  2 MiB
  const size_t HB  = (size_t)BH * SEQ * DK * 2;       // 16 MiB
  __bf16* xb  = (__bf16*)(ws);
  __bf16* wqb = (__bf16*)(ws + XB);
  __bf16* wkb = (__bf16*)(ws + XB + WB);
  __bf16* wvb = (__bf16*)(ws + XB + 2 * WB);
  __bf16* wob = (__bf16*)(ws + XB + 3 * WB);
  __bf16* qb  = (__bf16*)(ws + XB + 4 * WB);
  __bf16* kb  = (__bf16*)(ws + XB + 4 * WB + HB);
  __bf16* vtb = (__bf16*)(ws + XB + 4 * WB + 2 * HB);
  __bf16* aob = (__bf16*)(ws + XB + 4 * WB + 3 * HB);

  {
    const int nx4 = BL * DMODEL / 4;
    const int nw4 = DMODEL * DMODEL / 4;
    f32_to_bf16_kernel<<<nx4 / 256, 256, 0, stream>>>(x,  xb,  nx4);
    f32_to_bf16_kernel<<<nw4 / 256, 256, 0, stream>>>(Wq, wqb, nw4);
    f32_to_bf16_kernel<<<nw4 / 256, 256, 0, stream>>>(Wk, wkb, nw4);
    f32_to_bf16_kernel<<<nw4 / 256, 256, 0, stream>>>(Wv, wvb, nw4);
    f32_to_bf16_kernel<<<nw4 / 256, 256, 0, stream>>>(Wo, wob, nw4);
  }
  // Stage 1: 3 * 8192 32x32 tiles / 8 waves = 3072 blocks.
  qkv_gemm_kernel<<<3072, 256, 0, stream>>>(xb, wqb, wkb, wvb, bq, bk, bv,
                                            qb, kb, vtb);
  // Stage 2: 64 (b,h) * 128 strips; 128KB dynamic LDS.
  scores_softmax_kernel<<<BH * (SEQ / 16), 256, 16 * SEQ * sizeof(float), stream>>>(
      qb, kb, pattn);
  // Stage 3: 64 * 128 tiles / 8 = 1024 blocks.
  av_gemm_kernel<<<1024, 256, 0, stream>>>(pattn, vtb, aob);
  // Stage 4: 8192 tiles / 8 = 1024 blocks.
  oproj_gemm_kernel<<<1024, 256, 0, stream>>>(aob, wob, bo, out);
}
